// CausalSelfAttention_16801912062479
// MI455X (gfx1250) — compile-verified, matrix operand order checked by reading, not run on hardware
//
#include <hip/hip_runtime.h>
#include <hip/hip_bf16.h>
#include <stdint.h>

// ---------------------------------------------------------------------------
// CDNA5 (gfx1250) causal self-attention: bf16 WMMA + async-to-LDS pipelines.
// ---------------------------------------------------------------------------

typedef __bf16 bf16;
typedef __attribute__((ext_vector_type(16))) __bf16 v16bf;
typedef __attribute__((ext_vector_type(8)))  __bf16 v8bf;
typedef __attribute__((ext_vector_type(4)))  __bf16 v4bf;
typedef __attribute__((ext_vector_type(8)))  float  v8f;
typedef int v4i __attribute__((vector_size(4 * sizeof(int))));

#define DEV __device__ __forceinline__

static constexpr int B_  = 2;
static constexpr int T_  = 2048;
static constexpr int C_  = 1024;
static constexpr int H_  = 16;
static constexpr int D_  = 64;
static constexpr int M_  = B_ * T_;      // 4096 rows (B*T)
static constexpr int NQ_ = 3 * C_;       // 3072 qkv output cols

// GEMM block tile
static constexpr int BM = 128, BN = 64, BK = 64;

// ---- WMMA wrapper: D = A(16x32 bf16) x B(32x16 bf16) + C(16x16 f32) --------
DEV v8f wmma_bf16(v16bf a, v16bf b, v8f c) {
  return __builtin_amdgcn_wmma_f32_16x16x32_bf16(
      /*neg_a=*/false, a, /*neg_b=*/false, b,
      /*c_mod=*/(short)0, c, /*reuse_a=*/false, /*reuse_b=*/false);
}

// ---- Fragment loader (A 16x32, or B^T stored [N,K]) ------------------------
// ISA 7.12.2: lanes 0-15 (row M=lane) hold K=0..7 then K=16..23;
// lanes 16-31 hold K=8..15 then K=24..31.  Two contiguous 16B loads per lane.
DEV v16bf load_frag(const bf16* __restrict__ base, int ld) {
  const int lane = threadIdx.x & 31;
  const int row  = lane & 15;
  const int koff = (lane >> 4) << 3;          // 0 or 8
  const bf16* p = base + row * ld + koff;
  v8bf lo = *(const v8bf*)(p);
  v8bf hi = *(const v8bf*)(p + 16);
  v16bf f;
#pragma unroll
  for (int i = 0; i < 8; ++i) { f[i] = lo[i]; f[i + 8] = hi[i]; }
  return f;
}

// ---- CDNA5 async copy: global -> LDS, 16B, tracked by ASYNCcnt -------------
DEV void async_copy_b128(const bf16* gsrc, bf16* lds_dst) {
#if __has_builtin(__builtin_amdgcn_global_load_async_to_lds_b128)
  __builtin_amdgcn_global_load_async_to_lds_b128(
      (__attribute__((address_space(1))) v4i*)gsrc,
      (__attribute__((address_space(3))) v4i*)lds_dst,
      /*offset=*/0, /*cpol=*/0);
#else
  *(v8bf*)lds_dst = *(const v8bf*)gsrc;       // sync fallback
#endif
}

DEV void wait_async0() {
#if __has_builtin(__builtin_amdgcn_s_wait_asynccnt)
  __builtin_amdgcn_s_wait_asynccnt(0);
#else
  asm volatile("s_wait_asynccnt 0x0" ::: "memory");
#endif
}

// ---- Shared GEMM mainloop: C[128x64] += A[128xK] * B^T[64xK] ---------------
// 256 threads / 8 waves, waves in 4(M) x 2(N); each wave owns 32x32.
// Double-buffered LDS stages of BK=64 fed by async b128 copies.
DEV void gemm_mainloop(const bf16* __restrict__ Ag, const bf16* __restrict__ Bg,
                       long m0, long n0,
                       bf16 (*sA)[BM * BK], bf16 (*sB)[BN * BK],
                       v8f acc[2][2]) {
  const int tid  = threadIdx.x;
  const int wave = tid >> 5;
  const int wm   = wave >> 1;                 // 0..3 (M)
  const int wn   = wave & 1;                  // 0..1 (N)

  auto stage = [&](int buf, int k) {
    // A tile: 128x64 bf16 = 1024 16B-chunks, 4 per thread
#pragma unroll
    for (int c = 0; c < 4; ++c) {
      int chunk = tid + c * 256;
      int r = chunk >> 3, col = (chunk & 7) << 3;
      async_copy_b128(Ag + (m0 + r) * C_ + k + col, &sA[buf][r * BK + col]);
    }
    // B tile: 64x64 bf16 = 512 chunks, 2 per thread
#pragma unroll
    for (int c = 0; c < 2; ++c) {
      int chunk = tid + c * 256;
      int r = chunk >> 3, col = (chunk & 7) << 3;
      async_copy_b128(Bg + (n0 + r) * C_ + k + col, &sB[buf][r * BK + col]);
    }
  };

  stage(0, 0);
  for (int k = 0, it = 0; k < C_; k += BK, ++it) {
    const int buf = it & 1;
    wait_async0();
    __syncthreads();                          // staged buffer visible to all
    if (k + BK < C_) stage(buf ^ 1, k + BK);  // overlap next stage with math

    const bf16* Ab = &sA[buf][(wm * 32) * BK];
    const bf16* Bb = &sB[buf][(wn * 32) * BK];
#pragma unroll
    for (int ks = 0; ks < 2; ++ks) {          // two 32-wide K-steps
      v16bf a0 = load_frag(Ab + ks * 32, BK);
      v16bf a1 = load_frag(Ab + 16 * BK + ks * 32, BK);
      v16bf b0 = load_frag(Bb + ks * 32, BK);
      v16bf b1 = load_frag(Bb + 16 * BK + ks * 32, BK);
      acc[0][0] = wmma_bf16(a0, b0, acc[0][0]);
      acc[0][1] = wmma_bf16(a0, b1, acc[0][1]);
      acc[1][0] = wmma_bf16(a1, b0, acc[1][0]);
      acc[1][1] = wmma_bf16(a1, b1, acc[1][1]);
    }
  }
}

// ---- fp32 -> bf16 bulk convert ---------------------------------------------
__global__ void __launch_bounds__(256)
k_cvt_bf16(const float* __restrict__ in, bf16* __restrict__ out, int n4) {
  int i = blockIdx.x * blockDim.x + threadIdx.x;
  if (i < n4) {
    float4 v = ((const float4*)in)[i];
    v4bf o;
    o[0] = (bf16)v.x; o[1] = (bf16)v.y; o[2] = (bf16)v.z; o[3] = (bf16)v.w;
    ((v4bf*)out)[i] = o;
  }
}

// ---- [K,N] f32 -> [N,K] bf16 tiled transpose -------------------------------
__global__ void __launch_bounds__(256)
k_transpose_bf16(const float* __restrict__ w, bf16* __restrict__ wt,
                 int K, int N) {
  __shared__ bf16 tile[32][33];
  const int bx = blockIdx.x, by = blockIdx.y;
  const int tx = threadIdx.x & 31, ty = threadIdx.x >> 5;
#pragma unroll
  for (int i = 0; i < 4; ++i) {
    int kk = by * 32 + ty + i * 8;
    tile[ty + i * 8][tx] = (bf16)w[(long)kk * N + bx * 32 + tx];
  }
  __syncthreads();
#pragma unroll
  for (int i = 0; i < 4; ++i) {
    int nn = bx * 32 + ty + i * 8;
    wt[(long)nn * K + by * 32 + tx] = tile[tx][ty + i * 8];
  }
}

// ---- QKV GEMM + head scatter (q pre-scaled 1/8, v stored [BH,D,T]) ---------
__global__ void __launch_bounds__(256)
k_gemm_qkv(const bf16* __restrict__ xb, const bf16* __restrict__ wT,
           bf16* __restrict__ qb, bf16* __restrict__ kb,
           bf16* __restrict__ vT) {
  __shared__ __align__(16) bf16 sA[2][BM * BK];
  __shared__ __align__(16) bf16 sB[2][BN * BK];
  const int lane = threadIdx.x & 31;
  const int wave = threadIdx.x >> 5;
  const int wm = wave >> 1, wn = wave & 1;
  const long m0 = (long)blockIdx.x * BM;
  const long n0 = (long)blockIdx.y * BN;

  v8f acc[2][2] = {};
  gemm_mainloop(xb, wT, m0, n0, sA, sB, acc);

  const int lo = lane & 15, hi = lane >> 4;
  const int which = (int)(n0 >> 10);          // uniform (64 | 1024)
#pragma unroll
  for (int i = 0; i < 2; ++i)
#pragma unroll
    for (int j = 0; j < 2; ++j) {
      int col = (int)n0 + wn * 32 + j * 16 + lo;
      int c = col & (C_ - 1);
      int h = c >> 6, d = c & 63;
#pragma unroll
      for (int v = 0; v < 8; ++v) {
        int row = (int)m0 + wm * 32 + i * 16 + v + (hi << 3);
        int b = row >> 11, t = row & (T_ - 1);
        long bh = (long)(b * H_ + h);
        float val = acc[i][j][v];
        if (which == 0)      qb[(bh * T_ + t) * D_ + d] = (bf16)(val * 0.125f);
        else if (which == 1) kb[(bh * T_ + t) * D_ + d] = (bf16)val;
        else                 vT[(bh * D_ + d) * T_ + t] = (bf16)val;
      }
    }
}

// ---- Proj GEMM -> f32 output -----------------------------------------------
__global__ void __launch_bounds__(256)
k_gemm_proj(const bf16* __restrict__ yb, const bf16* __restrict__ wT,
            float* __restrict__ out) {
  __shared__ __align__(16) bf16 sA[2][BM * BK];
  __shared__ __align__(16) bf16 sB[2][BN * BK];
  const int lane = threadIdx.x & 31;
  const int wave = threadIdx.x >> 5;
  const int wm = wave >> 1, wn = wave & 1;
  const long m0 = (long)blockIdx.x * BM;
  const long n0 = (long)blockIdx.y * BN;

  v8f acc[2][2] = {};
  gemm_mainloop(yb, wT, m0, n0, sA, sB, acc);

  const int lo = lane & 15, hi = lane >> 4;
#pragma unroll
  for (int i = 0; i < 2; ++i)
#pragma unroll
    for (int j = 0; j < 2; ++j) {
      int col = (int)n0 + wn * 32 + j * 16 + lo;
#pragma unroll
      for (int v = 0; v < 8; ++v) {
        int row = (int)m0 + wm * 32 + i * 16 + v + (hi << 3);
        out[(long)row * C_ + col] = acc[i][j][v];
      }
    }
}

// ---- Flash attention: 1 wave per 16-row query tile, 32-key blocks ----------
__global__ void __launch_bounds__(128)
k_attn(const bf16* __restrict__ q, const bf16* __restrict__ k,
       const bf16* __restrict__ vt, bf16* __restrict__ yb) {
  __shared__ __align__(16) float Sb[4][16 * 32];
  __shared__ __align__(16) bf16  Pb[4][16 * 32];
  __shared__ float mS[4][16], lS[4][16], cS[4][16];

  const int wave  = threadIdx.x >> 5;
  const int lane  = threadIdx.x & 31;
  const int qt    = blockIdx.x * 4 + wave;
  const int bh    = blockIdx.y;
  const int qbase = qt * 16;

  const bf16* qh = q  + (long)bh * T_ * D_;
  const bf16* kh = k  + (long)bh * T_ * D_;
  const bf16* vh = vt + (long)bh * D_ * T_;

  v16bf aq0 = load_frag(qh + (long)qbase * D_,      D_);
  v16bf aq1 = load_frag(qh + (long)qbase * D_ + 32, D_);

  v8f o[4] = {};
  if (lane < 16) { mS[wave][lane] = -1e30f; lS[wave][lane] = 0.f; }
  asm volatile("s_wait_dscnt 0x0" ::: "memory");

  const int lo = lane & 15, hi = lane >> 4;

  for (int kb = 0; kb < qbase + 16; kb += 32) {
#pragma unroll
    for (int sub = 0; sub < 2; ++sub) {
      const int n0 = kb + sub * 16;
      v8f s = {};
      v16bf b0 = load_frag(kh + (long)n0 * D_,      D_);
      s = wmma_bf16(aq0, b0, s);
      v16bf b1 = load_frag(kh + (long)n0 * D_ + 32, D_);
      s = wmma_bf16(aq1, b1, s);
#pragma unroll
      for (int v = 0; v < 8; ++v) {
        int r    = v + (hi << 3);
        int colg = n0 + lo;
        float sv = s[v];
        if (colg > qbase + r) sv = -1e30f;    // causal mask
        Sb[wave][r * 32 + sub * 16 + lo] = sv;
      }
    }
    asm volatile("s_wait_dscnt 0x0" ::: "memory");

    if (lane < 16) {                          // online softmax, row = lane
      const int r = lane;
      float mOld = mS[wave][r];
      float mx = mOld;
#pragma unroll
      for (int c2 = 0; c2 < 32; ++c2) mx = fmaxf(mx, Sb[wave][r * 32 + c2]);
      float corr = __expf(mOld - mx);
      float sum = 0.f;
#pragma unroll
      for (int c2 = 0; c2 < 32; ++c2) {
        float p = __expf(Sb[wave][r * 32 + c2] - mx);
        Pb[wave][r * 32 + c2] = (bf16)p;
        sum += p;
      }
      lS[wave][r] = lS[wave][r] * corr + sum;
      mS[wave][r] = mx;
      cS[wave][r] = corr;
    }
    asm volatile("s_wait_dscnt 0x0" ::: "memory");

    float cf[8];
#pragma unroll
    for (int v = 0; v < 8; ++v) cf[v] = cS[wave][v + (hi << 3)];
#pragma unroll
    for (int dt = 0; dt < 4; ++dt)
#pragma unroll
      for (int v = 0; v < 8; ++v) o[dt][v] *= cf[v];

    v16bf ap = load_frag(&Pb[wave][0], 32);
#pragma unroll
    for (int dt = 0; dt < 4; ++dt) {
      v16bf bv = load_frag(vh + (long)(dt * 16) * T_ + kb, T_);
      o[dt] = wmma_bf16(ap, bv, o[dt]);
    }
  }

  float li[8];
#pragma unroll
  for (int v = 0; v < 8; ++v) li[v] = 1.0f / lS[wave][v + (hi << 3)];
  const int b = bh >> 4, h = bh & 15;
#pragma unroll
  for (int dt = 0; dt < 4; ++dt)
#pragma unroll
    for (int v = 0; v < 8; ++v) {
      int r = v + (hi << 3);
      int t = qbase + r;
      yb[((long)(b * T_ + t)) * C_ + h * D_ + dt * 16 + lo] =
          (bf16)(o[dt][v] * li[v]);
    }
}

// ---------------------------------------------------------------------------
extern "C" void kernel_launch(void* const* d_in, const int* in_sizes, int n_in,
                              void* d_out, int out_size, void* d_ws,
                              size_t ws_size, hipStream_t stream) {
  const float* x      = (const float*)d_in[0];   // [2,2048,1024]
  const float* w_qkv  = (const float*)d_in[1];   // [1024,3072]
  const float* w_proj = (const float*)d_in[2];   // [1024,1024]
  float* out = (float*)d_out;

  char* ws = (char*)d_ws;
  size_t off = 0;
  auto alloc = [&](size_t bytes) -> void* {
    void* p = ws + off;
    off += (bytes + 255) & ~(size_t)255;
    return p;
  };
  bf16* xb     = (bf16*)alloc((size_t)M_ * C_ * 2);
  bf16* wqkvT  = (bf16*)alloc((size_t)NQ_ * C_ * 2);
  bf16* wprojT = (bf16*)alloc((size_t)C_ * C_ * 2);
  bf16* qb     = (bf16*)alloc((size_t)B_ * H_ * T_ * D_ * 2);
  bf16* kb     = (bf16*)alloc((size_t)B_ * H_ * T_ * D_ * 2);
  bf16* vT     = (bf16*)alloc((size_t)B_ * H_ * D_ * T_ * 2);
  bf16* yb     = (bf16*)alloc((size_t)M_ * C_ * 2);

  {
    int n4 = M_ * C_ / 4;
    k_cvt_bf16<<<(n4 + 255) / 256, 256, 0, stream>>>(x, xb, n4);
  }
  k_transpose_bf16<<<dim3(NQ_ / 32, C_ / 32), 256, 0, stream>>>(w_qkv, wqkvT,
                                                                C_, NQ_);
  k_transpose_bf16<<<dim3(C_ / 32, C_ / 32), 256, 0, stream>>>(w_proj, wprojT,
                                                               C_, C_);
  k_gemm_qkv<<<dim3(M_ / BM, NQ_ / BN), 256, 0, stream>>>(xb, wqkvT, qb, kb,
                                                          vT);
  k_attn<<<dim3(T_ / 64, B_ * H_), 128, 0, stream>>>(qb, kb, vT, yb);
  k_gemm_proj<<<dim3(M_ / BM, C_ / BN), 256, 0, stream>>>(yb, wprojT, out);
}
